// OutProductMean_27161373180655
// MI455X (gfx1250) — matrix operand order repaired and checked by reading, not compiled, not run on hardware
//
#include <hip/hip_runtime.h>

// ---------------------------------------------------------------------------
// OuterProductMean on MI455X (gfx1250, wave32, WMMA bf16 16x16x32)
//
//  Z[i,j,o] = sum_{d,e} Wo[o, d*32+e] * sum_s L[s,i,d] * R[s,j,e]
//  with L/R = LayerNorm(M) @ Wa^T/Wb^T (+bias).
//
// Pipeline:
//  prep_kernel : LayerNorm + projections, emit bf16 Lt[(i*32+d), s],
//                Rt[(j*32+e), s]   (row-major, K=s contiguous)
//  conv_wo     : Wo f32 -> bf16 (row-major [o][1024], K=de contiguous)
//  fused_kernel: per (4i x 4j) tile:
//     GEMM1: O_blk(128x128, K=128), wave tile 2x4 (m x n) 16x16 WMMAs
//     GEMM2: Zt(16o x 16pairs, K=1024) = Wo x O_blk^T, dual accumulators
// ---------------------------------------------------------------------------

typedef __bf16 bf16_t;
typedef __attribute__((ext_vector_type(16))) __bf16 v16bf;
typedef __attribute__((ext_vector_type(8)))  __bf16 v8bf;
typedef __attribute__((ext_vector_type(8)))  float  v8f;

#define S_DIM 128
#define I_DIM 256
#define D_DIM 64
#define P_DIM 32
#define DO_DIM 128

// ------------------------- prep: LayerNorm + projections --------------------
// one wave per (s,i) vector; 8 waves / block; grid = 32768/8 = 4096 blocks
__global__ __launch_bounds__(256) void prep_kernel(
    const float* __restrict__ M, const float* __restrict__ gamma,
    const float* __restrict__ beta, const float* __restrict__ Wa,
    const float* __restrict__ ba, const float* __restrict__ Wb,
    const float* __restrict__ bb, bf16_t* __restrict__ Lt,
    bf16_t* __restrict__ Rt) {
  __shared__ float sh[8][D_DIM];
  const int wv = threadIdx.x >> 5;
  const int ln = threadIdx.x & 31;
  const int vi = blockIdx.x * 8 + wv;   // vector index over (s,i)
  const int s = vi >> 8;                // I_DIM == 256
  const int i = vi & 255;

  const float* x = M + (size_t)vi * D_DIM;
  float x0 = x[ln];
  float x1 = x[ln + 32];

  float sum = x0 + x1;
  for (int m = 16; m > 0; m >>= 1) sum += __shfl_xor(sum, m, 32);
  const float mean = sum * (1.0f / 64.0f);
  const float d0 = x0 - mean, d1 = x1 - mean;
  float ss = d0 * d0 + d1 * d1;
  for (int m = 16; m > 0; m >>= 1) ss += __shfl_xor(ss, m, 32);
  const float rstd = rsqrtf(ss * (1.0f / 64.0f) + 1e-5f);

  sh[wv][ln]      = d0 * rstd * gamma[ln]      + beta[ln];
  sh[wv][ln + 32] = d1 * rstd * gamma[ln + 32] + beta[ln + 32];
  __syncthreads();

  // lane ln == projection index p (P_DIM == 32 == wave size)
  float lv = ba[ln];
  float rv = bb[ln];
  const float* wa = Wa + ln * D_DIM;
  const float* wb = Wb + ln * D_DIM;
#pragma unroll 8
  for (int d = 0; d < D_DIM; ++d) {
    const float n = sh[wv][d];
    lv += n * wa[d];
    rv += n * wb[d];
  }
  Lt[((size_t)i * P_DIM + ln) * S_DIM + s] = (bf16_t)lv;
  Rt[((size_t)i * P_DIM + ln) * S_DIM + s] = (bf16_t)rv;
}

// ------------------------- Wo f32 -> bf16 ----------------------------------
__global__ __launch_bounds__(256) void conv_wo(const float* __restrict__ Wo,
                                               bf16_t* __restrict__ Wob) {
  const int base = (blockIdx.x * 256 + threadIdx.x) * 4;  // 128 blocks -> 131072
#pragma unroll
  for (int t = 0; t < 4; ++t) Wob[base + t] = (bf16_t)Wo[base + t];
}

// ----- fragment loaders (16-bit WMMA operand layouts, ISA 7.12.2) ----------
// A 16x32: lane half h -> K chunks [h*8, +8) and [16+h*8, +8)
static __device__ __forceinline__ v16bf load_afrag(const bf16_t* row, int h) {
  const v8bf a0 = *(const v8bf*)(row + h * 8);
  const v8bf a1 = *(const v8bf*)(row + 16 + h * 8);
  v16bf a;
#pragma unroll
  for (int t = 0; t < 8; ++t) { a[t] = a0[t]; a[t + 8] = a1[t]; }
  return a;
}
// B 32x16: lane half h -> K chunks [h*8, +8) and [16+h*8, +8) of this lane's column data
static __device__ __forceinline__ v16bf load_bfrag(const bf16_t* row, int h) {
  const v8bf b0 = *(const v8bf*)(row + h * 8);
  const v8bf b1 = *(const v8bf*)(row + 16 + h * 8);
  v16bf b;
#pragma unroll
  for (int t = 0; t < 8; ++t) { b[t] = b0[t]; b[t + 8] = b1[t]; }
  return b;
}

// ------------------------- fused GEMM1 + GEMM2 ------------------------------
// grid (64,64): 4x4 (i,j) tile per WG, 256 threads = 8 waves.
__global__ __launch_bounds__(256) void fused_kernel(
    const bf16_t* __restrict__ Lt, const bf16_t* __restrict__ Rt,
    const bf16_t* __restrict__ Wob, const float* __restrict__ bo,
    float* __restrict__ Z) {
  __shared__ bf16_t lA[128 * 128];  // Lt tile, later reused as bf16 O-block
  __shared__ bf16_t lB[128 * 128];  // Rt tile

  const int tid = threadIdx.x;
  const int wv = tid >> 5;
  const int ln = tid & 31;
  const int h = ln >> 4;        // lane half (0/1)
  const int c = ln & 15;        // row/col lane within half
  const int iBlk = blockIdx.x;
  const int jBlk = blockIdx.y;

  // ---- stage Lt/Rt tiles (each tile = contiguous 32 KB of global) ----
  {
    const v8bf* sA = (const v8bf*)(Lt + (size_t)iBlk * 128 * S_DIM);
    const v8bf* sB = (const v8bf*)(Rt + (size_t)jBlk * 128 * S_DIM);
    v8bf* dA = (v8bf*)lA;
    v8bf* dB = (v8bf*)lB;
#pragma unroll
    for (int t = 0; t < 8; ++t) {
      dA[tid + 256 * t] = sA[tid + 256 * t];
      dB[tid + 256 * t] = sB[tid + 256 * t];
    }
  }
  __syncthreads();

  // ---- GEMM1: O_blk(m=(ii,d) 128, n=(jj,e) 128, k=s 128) ----
  // Wave tile 2x4: m rows [32*(wv&3), +32), n cols [64*(wv>>2), +64).
  const int m0 = (wv & 3) * 32;
  const int n0 = (wv >> 2) * 64;

  v8f acc[2][4];
#pragma unroll
  for (int a = 0; a < 2; ++a)
#pragma unroll
    for (int n = 0; n < 4; ++n) acc[a][n] = 0.0f;

#pragma unroll
  for (int k = 0; k < 4; ++k) {  // K-steps of 32
    const v16bf af0 = load_afrag(&lA[(m0 + c) * 128 + k * 32], h);
    const v16bf af1 = load_afrag(&lA[(m0 + 16 + c) * 128 + k * 32], h);
    v16bf bf[4];
#pragma unroll
    for (int n = 0; n < 4; ++n)
      bf[n] = load_bfrag(&lB[(n0 + n * 16 + c) * 128 + k * 32], h);
#pragma unroll
    for (int n = 0; n < 4; ++n) {
      acc[0][n] = __builtin_amdgcn_wmma_f32_16x16x32_bf16(
          false, af0, false, bf[n], (short)0, acc[0][n], false, false);
      acc[1][n] = __builtin_amdgcn_wmma_f32_16x16x32_bf16(
          false, af1, false, bf[n], (short)0, acc[1][n], false, false);
    }
  }
  __syncthreads();

  // ---- spill O_blk as bf16 into lA (all GEMM1 LDS reads are done) ----
  // D layout: element v of acc[a][n] -> row (m0 + a*16 + v + 8h), col (n0 + n*16 + c)
#pragma unroll
  for (int a = 0; a < 2; ++a)
#pragma unroll
    for (int n = 0; n < 4; ++n)
#pragma unroll
      for (int v = 0; v < 8; ++v)
        lA[(m0 + a * 16 + v + 8 * h) * 128 + n0 + n * 16 + c] =
            (bf16_t)acc[a][n][v];
  __syncthreads();

  // ---- GEMM2: Zt(o=16 rows per wave, pairs=16, K=de 1024) ----
  // A = Wob rows (row-major, contiguous in de). B = O^T from LDS.
  // Dual accumulators to break the WMMA RAW chain; fully unrolled k-loop.
  v8f z0 = 0.0f, z1 = 0.0f;
  const bf16_t* wrow = Wob + (size_t)(wv * 16 + c) * 1024;
  const int p  = c;        // pair index 0..15
  const int ii = p >> 2;   // local i
  const int jj = p & 3;    // local j
#pragma unroll
  for (int kk = 0; kk < 32; kk += 2) {  // K-steps of 32; d == kk
    const v16bf wa0 = load_afrag(wrow + kk * 32, h);
    const v16bf ob0 = load_bfrag(&lA[(ii * 32 + kk) * 128 + jj * 32], h);
    z0 = __builtin_amdgcn_wmma_f32_16x16x32_bf16(
        false, wa0, false, ob0, (short)0, z0, false, false);
    const v16bf wa1 = load_afrag(wrow + (kk + 1) * 32, h);
    const v16bf ob1 = load_bfrag(&lA[(ii * 32 + kk + 1) * 128 + jj * 32], h);
    z1 = __builtin_amdgcn_wmma_f32_16x16x32_bf16(
        false, wa1, false, ob1, (short)0, z1, false, false);
  }

  // ---- scatter Zt -> Z[i,j,o] (+ bias) ----
  const int gi = iBlk * 4 + ii;
  const int gj = jBlk * 4 + jj;
  float* zrow = Z + ((size_t)gi * I_DIM + gj) * DO_DIM;
#pragma unroll
  for (int v = 0; v < 8; ++v) {
    const int o = wv * 16 + v + 8 * h;
    zrow[o] = z0[v] + z1[v] + bo[o];
  }
}

// ---------------------------------------------------------------------------
extern "C" void kernel_launch(void* const* d_in, const int* in_sizes, int n_in,
                              void* d_out, int out_size, void* d_ws, size_t ws_size,
                              hipStream_t stream) {
  const float* M     = (const float*)d_in[0];
  const float* gamma = (const float*)d_in[1];
  const float* beta  = (const float*)d_in[2];
  const float* Wa    = (const float*)d_in[3];
  const float* ba    = (const float*)d_in[4];
  const float* Wb    = (const float*)d_in[5];
  const float* bb    = (const float*)d_in[6];
  const float* Wo    = (const float*)d_in[7];
  const float* bo    = (const float*)d_in[8];
  float* Z = (float*)d_out;

  // workspace: Lt (2 MB) | Rt (2 MB) | Wob (256 KB)  -> 4.25 MB total
  bf16_t* Lt  = (bf16_t*)d_ws;
  bf16_t* Rt  = Lt + (size_t)I_DIM * P_DIM * S_DIM;
  bf16_t* Wob = Rt + (size_t)I_DIM * P_DIM * S_DIM;

  prep_kernel<<<dim3(4096), 256, 0, stream>>>(M, gamma, beta, Wa, ba, Wb, bb, Lt, Rt);
  conv_wo<<<dim3(128), 256, 0, stream>>>(Wo, Wob);
  fused_kernel<<<dim3(64, 64), 256, 0, stream>>>(Lt, Rt, Wob, bo, Z);
}
